// Block_14465449853283
// MI455X (gfx1250) — 16-bit and 8-bit matrix operand order repaired and checked by reading, compiled, not run on hardware
//
#include <hip/hip_runtime.h>
#include <hip/hip_bf16.h>
#include <stdint.h>

typedef __attribute__((ext_vector_type(8)))  int       v8i;
typedef __attribute__((ext_vector_type(8)))  float     v8f;
typedef __attribute__((ext_vector_type(8)))  _Float16  v8h;
typedef __attribute__((ext_vector_type(16))) _Float16  v16h;

#define TOKENS 8192
#define DIMC   384
#define DIMH3  1152
#define DIMFF  1536
#define SEQ    2048
#define NHEADS 6
#define HDIM   64

__device__ __forceinline__ v16h load_v16h(const _Float16* lo, const _Float16* hi) {
  v8h a = *(const v8h*)lo;
  v8h b = *(const v8h*)hi;
  v16h r;
#pragma unroll
  for (int i = 0; i < 8; ++i) { r[i] = a[i]; r[i + 8] = b[i]; }
  return r;
}

__global__ void zero4_kernel(float* p) {
  if (threadIdx.x < 4) p[threadIdx.x] = 0.0f;
}

__global__ void absmean_kernel(const float* __restrict__ w, int n, float* __restrict__ sum) {
  int idx = blockIdx.x * blockDim.x + threadIdx.x;
  float s = 0.0f;
  for (int i = idx; i < n; i += gridDim.x * blockDim.x) s += fabsf(w[i]);
#pragma unroll
  for (int off = 16; off >= 1; off >>= 1) s += __shfl_xor(s, off, 32);
  if ((threadIdx.x & 31) == 0) atomicAdd(sum, s);
}

__global__ void wquant_kernel(const float* __restrict__ w, int n,
                              const float* __restrict__ sum, float invcnt,
                              int8_t* __restrict__ wq) {
  int i = blockIdx.x * blockDim.x + threadIdx.x;
  if (i >= n) return;
  float mean = fmaxf(sum[0] * invcnt, 1e-5f);
  float qv = fminf(fmaxf(rintf(w[i] / mean), -1.0f), 1.0f);
  wq[i] = (int8_t)qv;
}

__global__ __launch_bounds__(256) void ln_quant_kernel(
    const float* __restrict__ x, const float* __restrict__ gam,
    const float* __restrict__ bet, int8_t* __restrict__ q,
    float* __restrict__ recip)
{
  int wave = threadIdx.x >> 5, lane = threadIdx.x & 31;
  int row = blockIdx.x * 8 + wave;
  const float* xr = x + (size_t)row * DIMC;
  float v[12];
#pragma unroll
  for (int i = 0; i < 3; ++i) {
    float4 t = *(const float4*)(xr + lane * 12 + i * 4);
    v[i*4+0] = t.x; v[i*4+1] = t.y; v[i*4+2] = t.z; v[i*4+3] = t.w;
  }
  float s = 0.0f;
#pragma unroll
  for (int i = 0; i < 12; ++i) s += v[i];
#pragma unroll
  for (int off = 16; off >= 1; off >>= 1) s += __shfl_xor(s, off, 32);
  float mu = s * (1.0f / DIMC);
  float vs = 0.0f;
#pragma unroll
  for (int i = 0; i < 12; ++i) { float d = v[i] - mu; vs += d * d; }
#pragma unroll
  for (int off = 16; off >= 1; off >>= 1) vs += __shfl_xor(vs, off, 32);
  float rstd = rsqrtf(vs * (1.0f / DIMC) + 1e-5f);
  float h[12]; float amax = 0.0f;
#pragma unroll
  for (int i = 0; i < 12; ++i) {
    int col = lane * 12 + i;
    h[i] = (v[i] - mu) * rstd * gam[col] + bet[col];
    amax = fmaxf(amax, fabsf(h[i]));
  }
#pragma unroll
  for (int off = 16; off >= 1; off >>= 1) amax = fmaxf(amax, __shfl_xor(amax, off, 32));
  float clipped = fmaxf(amax, 1e-5f);
  float scale = 128.0f / clipped;
  if (lane == 0) recip[row] = clipped * (1.0f / 128.0f);
  int8_t* qr = q + (size_t)row * DIMC + lane * 12;
#pragma unroll
  for (int i = 0; i < 3; ++i) {
    int packed = 0;
#pragma unroll
    for (int j = 0; j < 4; ++j) {
      float qq = fminf(fmaxf(rintf(h[i*4+j] * scale), -128.0f), 127.0f);
      packed |= ((int)qq & 0xff) << (8 * j);
    }
    *(int*)(qr + i * 4) = packed;
  }
}

__global__ __launch_bounds__(256) void act_quant_kernel(
    const float* __restrict__ in, int8_t* __restrict__ q,
    float* __restrict__ recip, int L, int do_gelu)
{
  int wave = threadIdx.x >> 5, lane = threadIdx.x & 31;
  int row = blockIdx.x * 8 + wave;
  const float* xr = in + (size_t)row * L;
  int cnt = L >> 5;
  float vals[48];
  float amax = 0.0f;
  for (int i = 0; i < cnt; ++i) {
    float v = xr[lane + i * 32];
    if (do_gelu) v = 0.5f * v * (1.0f + erff(v * 0.70710678118f));
    vals[i] = v;
    amax = fmaxf(amax, fabsf(v));
  }
#pragma unroll
  for (int off = 16; off >= 1; off >>= 1) amax = fmaxf(amax, __shfl_xor(amax, off, 32));
  float clipped = fmaxf(amax, 1e-5f);
  float scale = 128.0f / clipped;
  if (lane == 0) recip[row] = clipped * (1.0f / 128.0f);
  int8_t* qr = q + (size_t)row * L;
  for (int i = 0; i < cnt; ++i) {
    float qq = fminf(fmaxf(rintf(vals[i] * scale), -128.0f), 127.0f);
    qr[lane + i * 32] = (int8_t)qq;
  }
}

__global__ __launch_bounds__(256) void bitgemm_kernel(
    const int8_t* __restrict__ Aq, const float* __restrict__ arecip,
    const int8_t* __restrict__ Wq, const float* __restrict__ wsum, float winvcnt,
    const float* __restrict__ bias, const float* __restrict__ resid,
    float* __restrict__ out, int M, int N, int K)
{
  __shared__ int8_t lA[2][128 * 64];
  int tid  = threadIdx.x;
  int wave = tid >> 5, lane = tid & 31;
  int g = lane >> 4, ln = lane & 15;
  int m0 = blockIdx.x * 128, n0 = blockIdx.y * 128;
  int wm = (wave & 3) * 32, wn = (wave >> 2) * 64;

  int srow = tid >> 1, schunk = (tid & 1) * 32;
  uint64_t abase = (uint64_t)(uintptr_t)Aq;

  auto stage = [&](int buf, int kb) {
    uint32_t goff = (uint32_t)((m0 + srow) * K + kb + schunk);
    uint32_t l0   = (uint32_t)(uintptr_t)(&lA[buf][srow * 64 + schunk]);
    asm volatile("global_load_async_to_lds_b128 %0, %1, %2"
                 :: "v"(l0), "v"(goff), "s"(abase) : "memory");
    uint32_t goff1 = goff + 16;
    uint32_t l1    = l0 + 16;
    asm volatile("global_load_async_to_lds_b128 %0, %1, %2"
                 :: "v"(l1), "v"(goff1), "s"(abase) : "memory");
  };

  v8i acc[2][4] = {};

  stage(0, 0);
  asm volatile("s_wait_asynccnt 0" ::: "memory");
  __syncthreads();

  int nsteps = K >> 6;
  for (int step = 0; step < nsteps; ++step) {
    int kb  = step << 6;
    int cur = step & 1;
    if (step + 1 < nsteps) stage(cur ^ 1, kb + 64);

    v8i afrag[2];
#pragma unroll
    for (int mi = 0; mi < 2; ++mi) {
      const int8_t* base = &lA[cur][(wm + mi * 16 + ln) * 64 + g * 16];
      int2 p0 = *(const int2*)(base + 0);
      int2 p1 = *(const int2*)(base + 8);
      int2 p2 = *(const int2*)(base + 32);
      int2 p3 = *(const int2*)(base + 40);
      v8i a;
      a[0] = p0.x; a[1] = p0.y; a[2] = p1.x; a[3] = p1.y;
      a[4] = p2.x; a[5] = p2.y; a[6] = p3.x; a[7] = p3.y;
      afrag[mi] = a;
    }
#pragma unroll
    for (int ni = 0; ni < 4; ++ni) {
      const int8_t* wb = &Wq[(size_t)(n0 + wn + ni * 16 + ln) * K + kb + g * 16];
      __builtin_prefetch(wb + 64, 0, 0);
      int4 lo = *(const int4*)(wb);
      int4 hi = *(const int4*)(wb + 32);
      v8i b;
      b[0] = lo.x; b[1] = lo.y; b[2] = lo.z; b[3] = lo.w;
      b[4] = hi.x; b[5] = hi.y; b[6] = hi.z; b[7] = hi.w;
#pragma unroll
      for (int mi = 0; mi < 2; ++mi)
        acc[mi][ni] = __builtin_amdgcn_wmma_i32_16x16x64_iu8(
            true, afrag[mi], true, b, acc[mi][ni], false, false);
    }
    asm volatile("s_wait_asynccnt 0" ::: "memory");
    __syncthreads();
  }

  float sw = fmaxf(wsum[0] * winvcnt, 1e-5f);
#pragma unroll
  for (int mi = 0; mi < 2; ++mi)
#pragma unroll
    for (int ni = 0; ni < 4; ++ni)
#pragma unroll
      for (int r = 0; r < 8; ++r) {
        int m = m0 + wm + mi * 16 + r + 8 * g;
        int n = n0 + wn + ni * 16 + ln;
        float v = (float)acc[mi][ni][r] * arecip[m] * sw + bias[n];
        if (resid) v += resid[(size_t)m * N + n];
        out[(size_t)m * N + n] = v;
      }
}

__global__ void qkv_cast_kernel(const float* __restrict__ qkv,
                                _Float16* __restrict__ qf, _Float16* __restrict__ kf,
                                _Float16* __restrict__ vt)
{
  size_t idx = (size_t)blockIdx.x * blockDim.x + threadIdx.x;
  int col = (int)(idx % DIMH3);
  int tok = (int)(idx / DIMH3);
  int b = tok >> 11, nq = tok & 2047;
  int which = col / DIMC, c = col % DIMC;
  int h = c >> 6, d = c & 63;
  _Float16 hv = (_Float16)qkv[idx];
  size_t bh = (size_t)(b * NHEADS + h);
  if (which == 0)      qf[(bh * SEQ + nq) * HDIM + d] = hv;
  else if (which == 1) kf[(bh * SEQ + nq) * HDIM + d] = hv;
  else                 vt[(bh * HDIM + d) * SEQ + nq] = hv;
}

__global__ __launch_bounds__(256) void attn_kernel(
    const _Float16* __restrict__ qf, const _Float16* __restrict__ kf,
    const _Float16* __restrict__ vt, float* __restrict__ out)
{
  __shared__ _Float16 lP[8 * 16 * 64];
  int wave = threadIdx.x >> 5, lane = threadIdx.x & 31;
  int g = lane >> 4, ln = lane & 15;
  int bh = blockIdx.y;
  int b = bh / NHEADS, h = bh % NHEADS;
  int q0 = blockIdx.x * 128 + wave * 16;
  const _Float16* qb = qf + (size_t)bh * SEQ * HDIM;
  const _Float16* kb = kf + (size_t)bh * SEQ * HDIM;
  const _Float16* vb = vt + (size_t)bh * HDIM * SEQ;
  _Float16* pbuf = &lP[wave * 16 * 64];

  v16h qfrag[2];
#pragma unroll
  for (int c = 0; c < 2; ++c) {
    const _Float16* p = qb + (size_t)(q0 + ln) * HDIM + c * 32 + g * 8;
    qfrag[c] = load_v16h(p, p + 16);
  }

  float mrun[8], lrun[8];
  v8f oacc[4] = {};
#pragma unroll
  for (int r = 0; r < 8; ++r) { mrun[r] = -1e30f; lrun[r] = 0.0f; }

  for (int kv = 0; kv < SEQ; kv += 64) {
    v8f s[4];
#pragma unroll
    for (int nt = 0; nt < 4; ++nt) {
      v8f a = {};
#pragma unroll
      for (int c = 0; c < 2; ++c) {
        const _Float16* kp = kb + (size_t)(kv + nt * 16 + ln) * HDIM + c * 32 + g * 8;
        v16h bfrag = load_v16h(kp, kp + 16);
        a = __builtin_amdgcn_wmma_f32_16x16x32_f16(
            false, qfrag[c], false, bfrag, (short)0, a, false, false);
      }
#pragma unroll
      for (int r = 0; r < 8; ++r) a[r] *= 0.125f;
      s[nt] = a;
    }
#pragma unroll
    for (int r = 0; r < 8; ++r) {
      float mx = s[0][r];
#pragma unroll
      for (int nt = 1; nt < 4; ++nt) mx = fmaxf(mx, s[nt][r]);
#pragma unroll
      for (int off = 1; off <= 8; off <<= 1) mx = fmaxf(mx, __shfl_xor(mx, off, 32));
      float mnew = fmaxf(mrun[r], mx);
      float corr = __expf(mrun[r] - mnew);
      float ps = 0.0f;
#pragma unroll
      for (int nt = 0; nt < 4; ++nt) {
        float p = __expf(s[nt][r] - mnew);
        s[nt][r] = p;
        ps += p;
      }
#pragma unroll
      for (int off = 1; off <= 8; off <<= 1) ps += __shfl_xor(ps, off, 32);
      lrun[r] = lrun[r] * corr + ps;
      mrun[r] = mnew;
#pragma unroll
      for (int nt = 0; nt < 4; ++nt) oacc[nt][r] *= corr;
    }
#pragma unroll
    for (int nt = 0; nt < 4; ++nt)
#pragma unroll
      for (int r = 0; r < 8; ++r)
        pbuf[(r + 8 * g) * 64 + nt * 16 + ln] = (_Float16)s[nt][r];
    asm volatile("s_wait_dscnt 0" ::: "memory");
    v16h pfrag[2];
#pragma unroll
    for (int c = 0; c < 2; ++c) {
      const _Float16* pp = pbuf + (size_t)ln * 64 + c * 32 + g * 8;
      pfrag[c] = load_v16h(pp, pp + 16);
    }
#pragma unroll
    for (int nt = 0; nt < 4; ++nt) {
#pragma unroll
      for (int c = 0; c < 2; ++c) {
        const _Float16* vp = vb + (size_t)(nt * 16 + ln) * SEQ + kv + c * 32 + g * 8;
        v16h vfrag = load_v16h(vp, vp + 16);
        oacc[nt] = __builtin_amdgcn_wmma_f32_16x16x32_f16(
            false, pfrag[c], false, vfrag, (short)0, oacc[nt], false, false);
      }
    }
  }
#pragma unroll
  for (int nt = 0; nt < 4; ++nt)
#pragma unroll
    for (int r = 0; r < 8; ++r) {
      int qrow = q0 + r + 8 * g;
      int d = nt * 16 + ln;
      out[((size_t)(b * SEQ + qrow)) * DIMC + h * HDIM + d] = oacc[nt][r] / lrun[r];
    }
}

extern "C" void kernel_launch(void* const* d_in, const int* in_sizes, int n_in,
                              void* d_out, int out_size, void* d_ws, size_t ws_size,
                              hipStream_t stream)
{
  (void)in_sizes; (void)n_in; (void)out_size; (void)ws_size;
  const float* x      = (const float*)d_in[0];
  const float* ln1_w  = (const float*)d_in[1];
  const float* ln1_b  = (const float*)d_in[2];
  const float* ln2_w  = (const float*)d_in[3];
  const float* ln2_b  = (const float*)d_in[4];
  const float* qkv_w  = (const float*)d_in[5];
  const float* qkv_b  = (const float*)d_in[6];
  const float* proj_w = (const float*)d_in[7];
  const float* proj_b = (const float*)d_in[8];
  const float* fc1_w  = (const float*)d_in[9];
  const float* fc1_b  = (const float*)d_in[10];
  const float* fc2_w  = (const float*)d_in[11];
  const float* fc2_b  = (const float*)d_in[12];
  float* out = (float*)d_out;

  const int M = TOKENS, C = DIMC, H3 = DIMH3, FF = DIMFF;

  char* base = (char*)d_ws;
  size_t off = 0;
  auto alloc = [&](size_t bytes) -> char* {
    char* p = base + off;
    off += (bytes + 255) & ~(size_t)255;
    return p;
  };
  float*    wsums   = (float*)alloc(4 * sizeof(float));
  int8_t*   qkvw_q  = (int8_t*)alloc((size_t)H3 * C);
  int8_t*   projw_q = (int8_t*)alloc((size_t)C * C);
  int8_t*   fc1w_q  = (int8_t*)alloc((size_t)FF * C);
  int8_t*   fc2w_q  = (int8_t*)alloc((size_t)C * FF);
  int8_t*   xq      = (int8_t*)alloc((size_t)M * C);
  float*    recip1  = (float*)alloc(M * sizeof(float));
  char*     big1    = alloc((size_t)M * FF * sizeof(float));
  float*    qkvo    = (float*)big1;
  float*    h1      = (float*)big1;
  _Float16* qf16    = (_Float16*)alloc((size_t)M * C * 2);
  _Float16* kf16    = (_Float16*)alloc((size_t)M * C * 2);
  _Float16* vt16    = (_Float16*)alloc((size_t)M * C * 2);
  char*     big2    = alloc((size_t)M * FF);
  float*    attno   = (float*)big2;
  int8_t*   h1q     = (int8_t*)big2;
  int8_t*   oq      = (int8_t*)alloc((size_t)M * C);
  float*    recipo  = (float*)alloc(M * sizeof(float));
  float*    xa      = (float*)alloc((size_t)M * C * sizeof(float));
  int8_t*   hq      = (int8_t*)alloc((size_t)M * C);
  float*    reciph  = (float*)alloc(M * sizeof(float));
  float*    recipg  = (float*)alloc(M * sizeof(float));

  zero4_kernel<<<1, 32, 0, stream>>>(wsums);
  const float* ws_[4]  = { qkv_w, proj_w, fc1_w, fc2_w };
  int          wn_[4]  = { H3 * C, C * C, FF * C, C * FF };
  int8_t*      wq_[4]  = { qkvw_q, projw_q, fc1w_q, fc2w_q };
  for (int i = 0; i < 4; ++i)
    absmean_kernel<<<256, 256, 0, stream>>>(ws_[i], wn_[i], wsums + i);
  for (int i = 0; i < 4; ++i)
    wquant_kernel<<<(wn_[i] + 255) / 256, 256, 0, stream>>>(
        ws_[i], wn_[i], wsums + i, 1.0f / wn_[i], wq_[i]);

  ln_quant_kernel<<<M / 8, 256, 0, stream>>>(x, ln1_w, ln1_b, xq, recip1);

  bitgemm_kernel<<<dim3(M / 128, H3 / 128), 256, 0, stream>>>(
      xq, recip1, qkvw_q, wsums + 0, 1.0f / (H3 * C), qkv_b, nullptr, qkvo, M, H3, C);

  qkv_cast_kernel<<<(unsigned)(((size_t)M * H3) / 256), 256, 0, stream>>>(
      qkvo, qf16, kf16, vt16);

  attn_kernel<<<dim3(SEQ / 128, 4 * NHEADS), 256, 0, stream>>>(qf16, kf16, vt16, attno);

  act_quant_kernel<<<M / 8, 256, 0, stream>>>(attno, oq, recipo, C, 0);
  bitgemm_kernel<<<dim3(M / 128, C / 128), 256, 0, stream>>>(
      oq, recipo, projw_q, wsums + 1, 1.0f / (C * C), proj_b, x, xa, M, C, C);

  ln_quant_kernel<<<M / 8, 256, 0, stream>>>(xa, ln2_w, ln2_b, hq, reciph);
  bitgemm_kernel<<<dim3(M / 128, FF / 128), 256, 0, stream>>>(
      hq, reciph, fc1w_q, wsums + 2, 1.0f / (FF * C), fc1_b, nullptr, h1, M, FF, C);

  act_quant_kernel<<<M / 8, 256, 0, stream>>>(h1, h1q, recipg, FF, 1);
  bitgemm_kernel<<<dim3(M / 128, C / 128), 256, 0, stream>>>(
      h1q, recipg, fc2w_q, wsums + 3, 1.0f / (C * FF), fc2_b, xa, out, M, C, FF);
}
